// W4A8MoELayer_65936337928889
// MI455X (gfx1250) — 8-bit matrix operand order repaired and checked by reading, compiled, not run on hardware
//
#include <hip/hip_runtime.h>

typedef __attribute__((ext_vector_type(8))) int   v8i_t;
typedef __attribute__((ext_vector_type(8))) float v8f_t;

static constexpr int kNE   = 8;
static constexpr int kME   = 1024;
static constexpr int kK    = 2048;
static constexpr int kN    = 2048;
static constexpr int kG    = 128;
static constexpr int kM    = kNE * kME;
static constexpr int kKG   = kK / kG;

static constexpr int BLK_M = 128;
static constexpr int BLK_N = 128;
static constexpr int LDSS  = 144;

__global__ __launch_bounds__(256) void pack_int32_to_int8(
    const int* __restrict__ in, unsigned int* __restrict__ out, int n4)
{
  int i = blockIdx.x * blockDim.x + threadIdx.x;
  if (i >= n4) return;
  const int4 v = ((const int4*)in)[i];
  unsigned int b = ((unsigned)v.x & 0xFFu)
                 | (((unsigned)v.y & 0xFFu) << 8)
                 | (((unsigned)v.z & 0xFFu) << 16)
                 | (((unsigned)v.w & 0xFFu) << 24);
  out[i] = b;
}

__device__ __forceinline__ void async_copy_b128(const signed char* gsrc,
                                                unsigned lds_off)
{
  asm volatile("global_load_async_to_lds_b128 %0, %1, off"
               :
               : "v"(lds_off), "v"(gsrc)
               : "memory");
}

__device__ __forceinline__ void wait_async_le8()
{
  asm volatile("s_wait_asynccnt 0x8" ::: "memory");
}

__device__ __forceinline__ void wait_async_0()
{
  asm volatile("s_wait_asynccnt 0x0" ::: "memory");
}

__global__ __launch_bounds__(256) void w4a8_moe_gemm(
    const signed char* __restrict__ Apack,
    const signed char* __restrict__ Wpack,
    const float* __restrict__ per_tok,
    const float* __restrict__ per_chan,
    const float* __restrict__ gscales,
    float* __restrict__ out)
{
  __shared__ signed char sA[2][BLK_M * LDSS];
  __shared__ signed char sB[2][BLK_N * LDSS];

  const int n_blk = blockIdx.x * BLK_N;
  const int m_blk = blockIdx.y * BLK_M;
  const int e     = m_blk / kME;

  const int tid  = threadIdx.x;
  const int lane = tid & 31;
  const int wid  = tid >> 5;

  const int wm = (wid & 3) * 32;
  const int wn = (wid >> 2) * 64;

  const int st_row = tid >> 3;
  const int st_off = (tid & 7) * 16;

  const size_t wbase = (size_t)e * kN * kK;

  auto issue_group = [&](int g, int buf) {
#pragma unroll
    for (int p = 0; p < 4; ++p) {
      const int row = st_row + p * 32;
      async_copy_b128(
          Apack + (size_t)(m_blk + row) * kK + (size_t)g * kG + st_off,
          (unsigned)(size_t)&sA[buf][row * LDSS + st_off]);
      async_copy_b128(
          Wpack + wbase + (size_t)(n_blk + row) * kK + (size_t)g * kG + st_off,
          (unsigned)(size_t)&sB[buf][row * LDSS + st_off]);
    }
  };

  int colg[4];
#pragma unroll
  for (int ni = 0; ni < 4; ++ni)
    colg[ni] = n_blk + wn + ni * 16 + (lane & 15);

  v8f_t facc[2][4];
#pragma unroll
  for (int mi = 0; mi < 2; ++mi)
#pragma unroll
    for (int ni = 0; ni < 4; ++ni)
#pragma unroll
      for (int r = 0; r < 8; ++r)
        facc[mi][ni][r] = 0.0f;

  issue_group(0, 0);

  for (int g = 0; g < kKG; ++g) {
    if (g + 1 < kKG) {
      issue_group(g + 1, (g + 1) & 1);
      wait_async_le8();
    } else {
      wait_async_0();
    }
    __syncthreads();

    const signed char* curA = sA[g & 1];
    const signed char* curB = sB[g & 1];

    float gsc[4];
#pragma unroll
    for (int ni = 0; ni < 4; ++ni)
      gsc[ni] = gscales[(size_t)e * kN * kKG + (size_t)colg[ni] * kKG + g];

    v8i_t iacc[2][4];
#pragma unroll
    for (int mi = 0; mi < 2; ++mi)
#pragma unroll
      for (int ni = 0; ni < 4; ++ni)
#pragma unroll
        for (int r = 0; r < 8; ++r)
          iacc[mi][ni][r] = 0;

#pragma unroll
    for (int s = 0; s < 2; ++s) {
      v8i_t af[2];
#pragma unroll
      for (int mi = 0; mi < 2; ++mi) {
        const signed char* base =
            curA + (wm + mi * 16 + (lane & 15)) * LDSS + s * 64 + ((lane & 16) ? 16 : 0);
        int4 a0 = *(const int4*)(base +  0);
        int4 a1 = *(const int4*)(base + 32);
        af[mi][0] = a0.x; af[mi][1] = a0.y; af[mi][2] = a0.z; af[mi][3] = a0.w;
        af[mi][4] = a1.x; af[mi][5] = a1.y; af[mi][6] = a1.z; af[mi][7] = a1.w;
      }
      v8i_t bf[4];
#pragma unroll
      for (int ni = 0; ni < 4; ++ni) {
        const signed char* base =
            curB + (wn + ni * 16 + (lane & 15)) * LDSS + s * 64 + ((lane & 16) ? 16 : 0);
        int4 b0 = *(const int4*)(base +  0);
        int4 b1 = *(const int4*)(base + 32);
        bf[ni][0] = b0.x; bf[ni][1] = b0.y; bf[ni][2] = b0.z; bf[ni][3] = b0.w;
        bf[ni][4] = b1.x; bf[ni][5] = b1.y; bf[ni][6] = b1.z; bf[ni][7] = b1.w;
      }
#pragma unroll
      for (int mi = 0; mi < 2; ++mi)
#pragma unroll
        for (int ni = 0; ni < 4; ++ni)
          iacc[mi][ni] = __builtin_amdgcn_wmma_i32_16x16x64_iu8(
               true, af[mi],  true, bf[ni],
              iacc[mi][ni],  false,  false);
    }

#pragma unroll
    for (int mi = 0; mi < 2; ++mi)
#pragma unroll
      for (int ni = 0; ni < 4; ++ni)
#pragma unroll
        for (int r = 0; r < 8; ++r)
          facc[mi][ni][r] += gsc[ni] * (float)iacc[mi][ni][r];

    __syncthreads();
  }

#pragma unroll
  for (int mi = 0; mi < 2; ++mi) {
    const int rbase = m_blk + wm + mi * 16 + ((lane & 16) ? 8 : 0);
    float pts[8];
#pragma unroll
    for (int r = 0; r < 8; ++r) pts[r] = per_tok[rbase + r];
#pragma unroll
    for (int ni = 0; ni < 4; ++ni) {
      const float pcs = per_chan[(size_t)e * kN + colg[ni]];
#pragma unroll
      for (int r = 0; r < 8; ++r)
        out[(size_t)(rbase + r) * kN + colg[ni]] = facc[mi][ni][r] * pts[r] * pcs;
    }
  }
}

extern "C" void kernel_launch(void* const* d_in, const int* in_sizes, int n_in,
                              void* d_out, int out_size, void* d_ws, size_t ws_size,
                              hipStream_t stream) {
  const int*   a_q = (const int*)d_in[0];
  const int*   w_q = (const int*)d_in[1];
  const float* pt  = (const float*)d_in[2];
  const float* pc  = (const float*)d_in[3];
  const float* gs  = (const float*)d_in[4];
  float* out = (float*)d_out;

  signed char* Apack = (signed char*)d_ws;
  signed char* Wpack = Apack + (size_t)kM * kK;

  {
    int n4 = kM * kK / 4;
    pack_int32_to_int8<<<(n4 + 255) / 256, 256, 0, stream>>>(
        a_q, (unsigned int*)Apack, n4);
  }
  {
    int n4 = kNE * kN * kK / 4;
    pack_int32_to_int8<<<(n4 + 255) / 256, 256, 0, stream>>>(
        w_q, (unsigned int*)Wpack, n4);
  }

  dim3 grid(kN / BLK_N, kM / BLK_M);
  w4a8_moe_gemm<<<grid, 256, 0, stream>>>(Apack, Wpack, pt, pc, gs, out);
}
